// LSTM_encoder_6794638262639
// MI455X (gfx1250) — matrix operand order repaired and checked by reading, not compiled, not run on hardware
//
#include <hip/hip_runtime.h>

#define T_STEPS 2048
#define BATCH   64
#define DIM     256
#define HID     512
#define G4H     2048      // 4*HID
#define NWG     128
#define KC_X    8         // DIM/32
#define KC_H    16        // HID/32
#define KC_TOT  24        // (DIM+HID)/32

// dynamic LDS layout (bytes)
#define LDSW_BYTES (4 * KC_TOT * 512 * 2)          // 98304 : B-frags for 4 gates
#define LDSX_BYTES (KC_X * 32 * 32)                //  8192 : x A-frags (bf16)
#define LDSH_BYTES (KC_H * 32 * 32)                // 16384 : h A-frags (bf16)
#define DYN_LDS    (LDSW_BYTES + LDSX_BYTES + LDSH_BYTES)

typedef __attribute__((ext_vector_type(16))) __bf16        v16bf;
typedef __attribute__((ext_vector_type(8)))  float         v8f;
typedef __attribute__((ext_vector_type(4)))  unsigned int  u32x4;

union Frag32 {
  u32x4          q[2];
  v16bf          v;
  unsigned short s[16];
};

__device__ __forceinline__ unsigned short f32_to_bf16_rne(float f) {
  unsigned u = __builtin_bit_cast(unsigned, f);
  return (unsigned short)((u + 0x7fffu + ((u >> 16) & 1u)) >> 16);
}
// pack {hi16(f1), hi16(f0)} into one dword with a single v_perm_b32
__device__ __forceinline__ unsigned bfpack(unsigned f1, unsigned f0) {
  return __builtin_amdgcn_perm(f1, f0, 0x07060302u);
}
__device__ __forceinline__ float sigmoidf_(float x) {
  return 1.0f / (1.0f + __expf(-x));
}

// ---------------------------------------------------------------------------
// Prep 1: pack G = [W ; U]  (768 x 2048, f32) into bf16 WMMA B-fragment order.
// Fragment (nt, kc): 32 lanes x 16 bf16.  lane<16: n=lane,    k=kc*32 + e (+8 if e>=8)
//                                         lane>=16: n=lane-16, k=kc*32+8+e (+8 if e>=8)
// ---------------------------------------------------------------------------
__global__ void pack_weights(const float* __restrict__ W,
                             const float* __restrict__ U,
                             unsigned short* __restrict__ wfrag) {
  int tid = blockIdx.x * 256 + threadIdx.x;           // 0 .. 128*24*32-1
  if (tid >= 128 * KC_TOT * 32) return;
  int lane  = tid & 31;
  int kc    = (tid >> 5) % KC_TOT;
  int nt    = tid / (32 * KC_TOT);
  int n     = nt * 16 + (lane & 15);
  int kbase = kc * 32 + ((lane >> 4) ? 8 : 0);
  unsigned short* dst = wfrag + (size_t)tid * 16;
  #pragma unroll
  for (int e = 0; e < 16; ++e) {
    int k = kbase + e + ((e >= 8) ? 8 : 0);
    float v = (k < DIM) ? W[(size_t)k * G4H + n]
                        : U[(size_t)(k - DIM) * G4H + n];
    dst[e] = f32_to_bf16_rne(v);
  }
}

// Prep 2: zero h ping-pong buffers (h0 = 0) and per-step arrival counters.
__global__ void init_state(unsigned* __restrict__ hbuf_u32, int* __restrict__ cnt) {
  int i = blockIdx.x * 256 + threadIdx.x;
  if (i < 2 * BATCH * HID / 2) hbuf_u32[i] = 0u;   // 32768 dwords = both buffers
  if (i < T_STEPS)             cnt[i] = 0;
}

// ---------------------------------------------------------------------------
// Persistent cooperative LSTM: 128 WGs x 128 threads (4 waves).
// WG (bt, jt) owns batch rows [bt*16,+16) and h columns [jt*16,+16).
// Per step: waves cooperatively stage shared A-fragments (x in bf16 via
// v_perm packing, h passthrough) into LDS, then wave g accumulates the 16x16
// tile of gate g with 24 bf16 WMMAs fed entirely from LDS.
// ---------------------------------------------------------------------------
__global__ void __launch_bounds__(128, 1)
lstm_persistent(const float* __restrict__ x,
                const float* __restrict__ bias,
                const unsigned short* __restrict__ wfrag,
                unsigned short* __restrict__ hbuf,   // [2][64][512] bf16
                int* __restrict__ cnt,
                float* __restrict__ out) {
  __shared__ float lds_gates[4 * 256];
  __shared__ float lds_c[256];
  extern __shared__ char dynlds[];
  unsigned short* ldsW = (unsigned short*)dynlds;                    // 96 KB
  u32x4*          ldsX = (u32x4*)(dynlds + LDSW_BYTES);              //  8 KB
  u32x4*          ldsH = (u32x4*)(dynlds + LDSW_BYTES + LDSX_BYTES); // 16 KB

  const int tid  = threadIdx.x;
  const int lane = tid & 31;
  const int wave = tid >> 5;                         // gate id 0..3
  const int bt   = blockIdx.x & 3;
  const int jt   = blockIdx.x >> 2;

  // Stage this WG's weight fragments (4 gates x 24 k-chunks x 1KB) into LDS.
  {
    u32x4* d4 = (u32x4*)ldsW;
    #pragma unroll
    for (int g = 0; g < 4; ++g) {
      const u32x4* s4 = (const u32x4*)(wfrag + (size_t)(g * 32 + jt) * (KC_TOT * 512));
      for (int i = tid; i < 1536; i += 128) d4[g * 1536 + i] = s4[i];
    }
  }
  lds_c[tid] = 0.0f;
  lds_c[tid + 128] = 0.0f;
  __syncthreads();

  const int m   = lane & 15;        // batch row within tile (A-matrix M) == C-layout N
  const int hi  = lane >> 4;        // lane half selects K sub-groups
  const int row = bt * 16 + m;
  const unsigned* xrow = (const unsigned*)x + (size_t)row * T_STEPS * DIM;
  const float bw = bias[wave * HID + jt * 16 + m];   // gate bias, per-lane column

  for (int t = 0; t < T_STEPS; ++t) {
    if (t > 0) {
      if (tid == 0) {
        while (__hip_atomic_load(&cnt[t - 1], __ATOMIC_ACQUIRE,
                                 __HIP_MEMORY_SCOPE_AGENT) < NWG)
          __builtin_amdgcn_s_sleep(1);
      }
      __syncthreads();
      __threadfence();               // acquire: see other WGPs' fresh h
    }

    const unsigned* xt = xrow + (size_t)t * DIM;
    const unsigned short* hsrc =
        hbuf + (size_t)(t & 1) * (BATCH * HID) + (size_t)row * HID;

    if (wave == 0 && t + 1 < T_STEPS)   // pull next step's x toward the caches
      __builtin_prefetch((const void*)(xt + DIM + lane * 8), 0, 1);

    // ---- cooperative A-fragment staging: wave w -> x chunks {2w,2w+1},
    //      h chunks {4w..4w+3}; lane pattern identical across waves ----
    #pragma unroll
    for (int cc = 0; cc < 2; ++cc) {
      int kc = wave * 2 + cc;
      int k0 = kc * 32 + hi * 8;                     // K groups [k0,+8) & [k0+16,+8)
      u32x4 w0 = *(const u32x4*)(xt + k0);
      u32x4 w1 = *(const u32x4*)(xt + k0 + 4);
      u32x4 w2 = *(const u32x4*)(xt + k0 + 16);
      u32x4 w3 = *(const u32x4*)(xt + k0 + 20);
      u32x4 p0, p1;
      p0.x = bfpack(w0.y, w0.x);  p0.y = bfpack(w0.w, w0.z);
      p0.z = bfpack(w1.y, w1.x);  p0.w = bfpack(w1.w, w1.z);
      p1.x = bfpack(w2.y, w2.x);  p1.y = bfpack(w2.w, w2.z);
      p1.z = bfpack(w3.y, w3.x);  p1.w = bfpack(w3.w, w3.z);
      u32x4* xd = ldsX + ((size_t)kc * 32 + lane) * 2;
      xd[0] = p0;
      xd[1] = p1;
    }
    #pragma unroll
    for (int cc = 0; cc < 4; ++cc) {
      int kc = wave * 4 + cc;
      int k0 = kc * 32 + hi * 8;
      u32x4 h0 = *(const u32x4*)(hsrc + k0);         // 8 bf16
      u32x4 h1 = *(const u32x4*)(hsrc + k0 + 16);
      u32x4* hd = ldsH + ((size_t)kc * 32 + lane) * 2;
      hd[0] = h0;
      hd[1] = h1;
    }
    __syncthreads();

    v8f acc;
    #pragma unroll
    for (int r = 0; r < 8; ++r) acc[r] = bw;         // bias pre-folded into C

    // ---- x_t @ W  (8 chunks), all operands from LDS ----
    #pragma unroll
    for (int kc = 0; kc < KC_X; ++kc) {
      Frag32 a, bf;
      const u32x4* xs = ldsX + ((size_t)kc * 32 + lane) * 2;
      a.q[0] = xs[0];  a.q[1] = xs[1];
      const u32x4* wl = (const u32x4*)ldsW + ((size_t)(wave * KC_TOT + kc) * 32 + lane) * 2;
      bf.q[0] = wl[0]; bf.q[1] = wl[1];
      acc = __builtin_amdgcn_wmma_f32_16x16x32_bf16(false, a.v, false, bf.v,
                                                    (short)0, acc, false, false);
    }
    // ---- h_{t-1} @ U  (16 chunks), all operands from LDS ----
    #pragma unroll
    for (int kc = 0; kc < KC_H; ++kc) {
      Frag32 a, bf;
      const u32x4* hs = ldsH + ((size_t)kc * 32 + lane) * 2;
      a.q[0] = hs[0];  a.q[1] = hs[1];
      const u32x4* wl =
          (const u32x4*)ldsW + ((size_t)(wave * KC_TOT + KC_X + kc) * 32 + lane) * 2;
      bf.q[0] = wl[0]; bf.q[1] = wl[1];
      acc = __builtin_amdgcn_wmma_f32_16x16x32_bf16(false, a.v, false, bf.v,
                                                    (short)0, acc, false, false);
    }

    // C layout: lanes 0-15 -> M=r, lanes 16-31 -> M=r+8; N = lane%16
    #pragma unroll
    for (int r = 0; r < 8; ++r)
      lds_gates[wave * 256 + (r + hi * 8) * 16 + m] = acc[r];
    __syncthreads();

    // ---- elementwise LSTM cell: 2 elements per thread ----
    unsigned short* hdst = hbuf + (size_t)((t + 1) & 1) * (BATCH * HID);
    #pragma unroll
    for (int e2 = 0; e2 < 2; ++e2) {
      int e  = tid * 2 + e2;
      int em = e >> 4, en = e & 15;
      float iv = sigmoidf_(lds_gates[e]);
      float fv = sigmoidf_(lds_gates[256 + e]);
      float gv = tanhf    (lds_gates[512 + e]);
      float ov = sigmoidf_(lds_gates[768 + e]);
      float cn = fv * lds_c[e] + iv * gv;
      float hn = ov * tanhf(cn);
      lds_c[e] = cn;
      int grow = bt * 16 + em, gcol = jt * 16 + en;
      hdst[(size_t)grow * HID + gcol] = f32_to_bf16_rne(hn);
      if (t == T_STEPS - 1) {
        out[(size_t)grow * HID + gcol]               = hn;   // h_T
        out[BATCH * HID + (size_t)grow * HID + gcol] = cn;   // c_T
      }
    }

    __threadfence();                 // release: make h stores agent-visible
    __syncthreads();
    if (tid == 0)
      __hip_atomic_fetch_add(&cnt[t], 1, __ATOMIC_RELEASE,
                             __HIP_MEMORY_SCOPE_AGENT);
  }
}

// ---------------------------------------------------------------------------
extern "C" void kernel_launch(void* const* d_in, const int* in_sizes, int n_in,
                              void* d_out, int out_size, void* d_ws, size_t ws_size,
                              hipStream_t stream) {
  const float* x = (const float*)d_in[0];   // [64, 2048, 256]
  const float* W = (const float*)d_in[1];   // [256, 2048]
  const float* U = (const float*)d_in[2];   // [512, 2048]
  const float* b = (const float*)d_in[3];   // [2048]
  float* out = (float*)d_out;               // h_T (64x512) ++ c_T (64x512)

  char* ws = (char*)d_ws;
  unsigned short* wfrag = (unsigned short*)ws;                      // 3,145,728 B
  unsigned short* hbuf  = (unsigned short*)(ws + 3145728);          //   131,072 B
  int*            cnt   = (int*)(ws + 3145728 + 131072);            //     8,192 B

  (void)in_sizes; (void)n_in; (void)out_size; (void)ws_size;

  pack_weights<<<384, 256, 0, stream>>>(W, U, wfrag);
  init_state<<<128, 256, 0, stream>>>((unsigned*)hbuf, cnt);

  (void)hipFuncSetAttribute((const void*)lstm_persistent,
                            hipFuncAttributeMaxDynamicSharedMemorySize, DYN_LDS);
  lstm_persistent<<<NWG, 128, DYN_LDS, stream>>>(x, b, wfrag, hbuf, cnt, out);
}
